// Transformer_66838281061131
// MI455X (gfx1250) — matrix operand order repaired and checked by reading, not compiled, not run on hardware
//
#include <hip/hip_runtime.h>
#include <hip/hip_bf16.h>

// ---------------- model constants ----------------
#define Bn     4
#define Cc     3
#define Wn     30720
#define Pp     30
#define NPn    1024
#define Ss     1025
#define Ee     384
#define Hh     6
#define HDn    64
#define MLPD   1536
#define DEPTHn 12
#define NCn    10
#define MROWS  (Bn * Ss)     // 4100
#define MPAD   4160          // 65 * 64  (GEMM M padding)
#define SPAD   1056          // 33 * 32  (attention seq padding)

// Try CDNA5 async global->LDS staging if the toolchain exposes it.
#ifndef ATHENA_TRY_ASYNC
#define ATHENA_TRY_ASYNC 1
#endif
#if ATHENA_TRY_ASYNC
#if __has_builtin(__builtin_amdgcn_global_load_async_to_lds_b128)
#define HAVE_ASYNC_LDS 1
#endif
#endif

typedef __attribute__((ext_vector_type(16))) _Float16 v16h;
typedef __attribute__((ext_vector_type(8)))  _Float16 h8;
typedef __attribute__((ext_vector_type(4)))  _Float16 h4;
typedef __attribute__((ext_vector_type(8)))  float    v8f;
typedef __attribute__((ext_vector_type(4)))  float    f4;
typedef __attribute__((ext_vector_type(4)))  int      v4i;

typedef __attribute__((address_space(1))) v4i g_v4i;   // global int4
typedef __attribute__((address_space(3))) v4i l_v4i;   // LDS int4

union AFrag { v16h v; h8 h[2]; };

__device__ __forceinline__ v8f wmma_f16(v16h a, v16h b, v8f c) {
    // D = A(16x32 f16) * B(32x16 f16) + C(16x16 f32)
    return __builtin_amdgcn_wmma_f32_16x16x32_f16(
        /*neg_a=*/false, a, /*neg_b=*/false, b,
        /*c_mod=*/(short)0, c, /*reuse_a=*/false, /*reuse_b=*/false);
}

#ifdef HAVE_ASYNC_LDS
__device__ __forceinline__ void wait_async0() {
#if __has_builtin(__builtin_amdgcn_s_wait_asynccnt)
    __builtin_amdgcn_s_wait_asynccnt(0);
#else
    asm volatile("s_wait_asynccnt 0x0" ::: "memory");
#endif
}
#endif

__device__ __forceinline__ float redmax16(float t) {
#pragma unroll
    for (int off = 1; off < 16; off <<= 1) t = fmaxf(t, __shfl_xor(t, off, 32));
    return t;
}
__device__ __forceinline__ float redsum16(float t) {
#pragma unroll
    for (int off = 1; off < 16; off <<= 1) t += __shfl_xor(t, off, 32);
    return t;
}
__device__ __forceinline__ float redsum32(float t) {
#pragma unroll
    for (int off = 1; off < 32; off <<= 1) t += __shfl_xor(t, off, 32);
    return t;
}

// ---------------- patch embedding + positional encoding ----------------
__global__ __launch_bounds__(256) void patch_embed(
    const float* __restrict__ x, const float* __restrict__ cw,
    const float* __restrict__ cb, float* __restrict__ h)
{
    int idx = blockIdx.x * 256 + threadIdx.x;          // [0, 4096*384)
    int e = idx % Ee;
    int m = idx / Ee;
    int b = m >> 10;
    int n = m & 1023;
    const float* xp = x + (size_t)b * Cc * Wn + (size_t)n * Pp;
    const float* wp = cw + (size_t)e * Cc * Pp;
    float acc = cb[e];
#pragma unroll
    for (int c = 0; c < Cc; ++c)
#pragma unroll
        for (int p = 0; p < Pp; ++p)
            acc += xp[(size_t)c * Wn + p] * wp[c * Pp + p];
    // sinusoidal PE, position n+1
    float pos = (float)(n + 1);
    float ang = pos * __expf(-(float)(e & ~1) * (9.210340371976184f / (float)Ee));
    float pe  = (e & 1) ? __cosf(ang) : __sinf(ang);
    h[((size_t)b * Ss + 1 + n) * Ee + e] = acc + pe;
}

__global__ __launch_bounds__(128) void cls_pe(
    const float* __restrict__ cls, float* __restrict__ h)
{
    int e = blockIdx.x * 128 + threadIdx.x;
    if (e >= Ee) return;
    float pe  = (e & 1) ? 1.0f : 0.0f;   // pos=0: sin(0)=0, cos(0)=1
    float val = cls[e] + pe;
#pragma unroll
    for (int b = 0; b < Bn; ++b)
        h[((size_t)b * Ss) * Ee + e] = val;
}

// ---------------- layernorm: fp32 in -> f16 out ----------------
__global__ __launch_bounds__(128) void ln_kernel(
    const float* __restrict__ x, const float* __restrict__ w,
    const float* __restrict__ b, _Float16* __restrict__ y)
{
    int r = blockIdx.x;
    int t = threadIdx.x;
    const float* xr = x + (size_t)r * Ee;
    float v0 = xr[t], v1 = xr[t + 128], v2 = xr[t + 256];
    float s  = v0 + v1 + v2;
    float s2 = v0 * v0 + v1 * v1 + v2 * v2;
    s  = redsum32(s);
    s2 = redsum32(s2);
    __shared__ float red[8];
    int wave = t >> 5, lane = t & 31;
    if (lane == 0) { red[wave] = s; red[4 + wave] = s2; }
    __syncthreads();
    s  = red[0] + red[1] + red[2] + red[3];
    s2 = red[4] + red[5] + red[6] + red[7];
    float mu  = s * (1.0f / Ee);
    float var = s2 * (1.0f / Ee) - mu * mu;
    float rs  = rsqrtf(var + 1e-5f);
    _Float16* yr = y + (size_t)r * Ee;
    yr[t]       = (_Float16)((v0 - mu) * rs * w[t]       + b[t]);
    yr[t + 128] = (_Float16)((v1 - mu) * rs * w[t + 128] + b[t + 128]);
    yr[t + 256] = (_Float16)((v2 - mu) * rs * w[t + 256] + b[t + 256]);
}

// ---------------- generic WMMA GEMM: Out = epi(A_f16[M,K] @ W_f32[N,K]^T) ----------------
// 128 threads = 4 waves; workgroup tile 64(M) x 64(N); per-wave 16(M) x 64(N).
enum { EPI_QKV = 0, EPI_RESID = 1, EPI_GELU = 2 };

template <int EPI>
__global__ __launch_bounds__(128) void gemm_f16w32(
    const _Float16* __restrict__ A, const float* __restrict__ W,
    const float* __restrict__ bias, int K, int N,
    float* __restrict__ outF, _Float16* __restrict__ out16,
    _Float16* __restrict__ qb, _Float16* __restrict__ kb,
    _Float16* __restrict__ vtb)
{
    __shared__ _Float16 sA[64 * 40];   // 64 rows x 32 k, stride 40 (16B-aligned rows)
    __shared__ _Float16 sB[64 * 40];
    const int t    = threadIdx.x;
    const int wave = t >> 5;
    const int lane = t & 31;
    const int lm   = lane & 15;
    const int kh   = lane >> 4;
    const int n0   = blockIdx.x * 64;
    const int m0   = blockIdx.y * 64;

    v8f c[4] = {};

    for (int k0 = 0; k0 < K; k0 += 32) {
        // stage A tile: 64x32 f16, 2 x 16B per thread
        {
            int cid = t * 2;
#pragma unroll
            for (int i = 0; i < 2; ++i, ++cid) {
                int r = cid >> 2, off = (cid & 3) * 8;
#ifdef HAVE_ASYNC_LDS
                // CDNA5 async global->LDS copy (ASYNCcnt tracked); A is already f16
                __builtin_amdgcn_global_load_async_to_lds_b128(
                    (g_v4i*)(A + (size_t)(m0 + r) * K + k0 + off),
                    (l_v4i*)(sA + r * 40 + off),
                    0, 0);
#else
                h8 val = *(const h8*)(A + (size_t)(m0 + r) * K + k0 + off);
                *(h8*)(sA + r * 40 + off) = val;
#endif
            }
        }
        // stage W tile: 64x32 fp32 -> f16, 4 x float4 per thread (must pass through VGPRs)
        {
            int cid = t * 4;
#pragma unroll
            for (int i = 0; i < 4; ++i, ++cid) {
                int r = cid >> 3, off = (cid & 7) * 4;
                f4 wv = *(const f4*)(W + (size_t)(n0 + r) * K + k0 + off);
                h4 hv;
                hv[0] = (_Float16)wv[0]; hv[1] = (_Float16)wv[1];
                hv[2] = (_Float16)wv[2]; hv[3] = (_Float16)wv[3];
                *(h4*)(sB + r * 40 + off) = hv;
            }
        }
#ifdef HAVE_ASYNC_LDS
        wait_async0();
#endif
        __syncthreads();
        // speculative prefetch of next W k-slice into cache
        if (k0 + 32 < K)
            __builtin_prefetch(W + (size_t)(n0 + (t >> 1)) * K + k0 + 32 + (t & 1) * 16, 0, 0);

        // A fragment (half 0 -> K 0..15, half 1 -> K 16..31)
        AFrag a;
        a.h[0] = *(const h8*)(sA + (wave * 16 + lm) * 40 + kh * 16);
        a.h[1] = *(const h8*)(sA + (wave * 16 + lm) * 40 + kh * 16 + 8);
#pragma unroll
        for (int f = 0; f < 4; ++f) {
            // B fragment (lanes 0-15: K 0..15, lanes 16-31: K 16..31, per column n)
            AFrag bfr;
            const _Float16* bp = sB + (f * 16 + lm) * 40 + kh * 16;
            bfr.h[0] = *(const h8*)bp;
            bfr.h[1] = *(const h8*)(bp + 8);
            c[f] = wmma_f16(a.v, bfr.v, c[f]);
        }
        __syncthreads();
    }

    // epilogue: C layout -> lane holds column n = f*16+lm, rows m = v + 8*kh
#pragma unroll
    for (int f = 0; f < 4; ++f) {
        int n = n0 + f * 16 + lm;
        float bv = (EPI == EPI_QKV) ? 0.0f : bias[n];
#pragma unroll
        for (int v = 0; v < 8; ++v) {
            int m = m0 + wave * 16 + v + kh * 8;
            float val = c[f][v] + bv;
            if (EPI == EPI_RESID) {
                outF[(size_t)m * N + n] += val;
            } else if (EPI == EPI_GELU) {
                float g = 0.5f * val * (1.0f + erff(val * 0.70710678118654752f));
                out16[(size_t)m * N + n] = (_Float16)g;
            } else { // EPI_QKV: split into q (pre-scaled), k, v-transposed
                if (m < MROWS) {
                    int b = m / Ss, s = m - b * Ss;
                    int which = n / Ee;
                    int rem   = n - which * Ee;
                    int hh    = rem >> 6, d = rem & 63;
                    size_t bh = (size_t)(b * Hh + hh);
                    if (which == 0)
                        qb[(bh * SPAD + s) * HDn + d] = (_Float16)(val * 0.125f); // 1/sqrt(64)
                    else if (which == 1)
                        kb[(bh * SPAD + s) * HDn + d] = (_Float16)val;
                    else
                        vtb[(bh * HDn + d) * SPAD + s] = (_Float16)val;
                }
            }
        }
    }
}

// ---------------- flash attention: 1 wave per 16 q-rows per (b,h) ----------------
__global__ __launch_bounds__(32) void attn_kernel(
    const _Float16* __restrict__ qb, const _Float16* __restrict__ kb,
    const _Float16* __restrict__ vtb, _Float16* __restrict__ o16)
{
    __shared__ _Float16 sP[16 * 40];
    const int lane = threadIdx.x;
    const int lm = lane & 15, kh = lane >> 4;
    const int qrow0 = blockIdx.x * 16;
    const int bh = blockIdx.y;
    const int b = bh / Hh, hh = bh % Hh;

    const _Float16* qbase = qb  + (size_t)bh * SPAD * HDn;
    const _Float16* kbase = kb  + (size_t)bh * SPAD * HDn;
    const _Float16* vbase = vtb + (size_t)bh * HDn * SPAD;

    // Q fragments for d-chunks [0,32) and [32,64); live in regs all kernel
    AFrag aq0, aq1;
    {
        const _Float16* qp = qbase + (size_t)(qrow0 + lm) * HDn;
        aq0.h[0] = *(const h8*)(qp + kh * 16);
        aq0.h[1] = *(const h8*)(qp + kh * 16 + 8);
        aq1.h[0] = *(const h8*)(qp + 32 + kh * 16);
        aq1.h[1] = *(const h8*)(qp + 32 + kh * 16 + 8);
    }

    v8f o0 = {}, o1 = {}, o2 = {}, o3 = {};
    float mrow[8], lrow[8];
#pragma unroll
    for (int v = 0; v < 8; ++v) { mrow[v] = -1e30f; lrow[v] = 0.0f; }

    for (int kt = 0; kt < SPAD; kt += 32) {
        // ---- scores S[16 x 32] = Q(16x64) @ K^T ----
        v8f sc0 = {}, sc1 = {};
        {
            const _Float16* kp = kbase + (size_t)(kt + lm) * HDn;
            AFrag b0, b1;
            b0.h[0] = *(const h8*)(kp + kh * 16);
            b0.h[1] = *(const h8*)(kp + kh * 16 + 8);
            b1.h[0] = *(const h8*)(kp + 32 + kh * 16);
            b1.h[1] = *(const h8*)(kp + 32 + kh * 16 + 8);
            sc0 = wmma_f16(aq0.v, b0.v, sc0);
            sc0 = wmma_f16(aq1.v, b1.v, sc0);
        }
        {
            const _Float16* kp = kbase + (size_t)(kt + 16 + lm) * HDn;
            AFrag b0, b1;
            b0.h[0] = *(const h8*)(kp + kh * 16);
            b0.h[1] = *(const h8*)(kp + kh * 16 + 8);
            b1.h[0] = *(const h8*)(kp + 32 + kh * 16);
            b1.h[1] = *(const h8*)(kp + 32 + kh * 16 + 8);
            sc1 = wmma_f16(aq0.v, b0.v, sc1);
            sc1 = wmma_f16(aq1.v, b1.v, sc1);
        }
        // mask out-of-range keys
        if (kt + lm >= Ss) {
#pragma unroll
            for (int v = 0; v < 8; ++v) sc0[v] = -1e30f;
        }
        if (kt + 16 + lm >= Ss) {
#pragma unroll
            for (int v = 0; v < 8; ++v) sc1[v] = -1e30f;
        }
        // ---- online softmax update (row stats reduced across 16-lane halves) ----
#pragma unroll
        for (int v = 0; v < 8; ++v) {
            float tmax = redmax16(fmaxf(sc0[v], sc1[v]));
            float mn   = fmaxf(mrow[v], tmax);
            float corr = __expf(mrow[v] - mn);
            float p0   = __expf(sc0[v] - mn);
            float p1   = __expf(sc1[v] - mn);
            float rs   = redsum16(p0 + p1);
            lrow[v] = lrow[v] * corr + rs;
            mrow[v] = mn;
            o0[v] *= corr; o1[v] *= corr; o2[v] *= corr; o3[v] *= corr;
            sc0[v] = p0; sc1[v] = p1;
        }
        // ---- P (C-layout) -> LDS -> A-fragment layout ----
#pragma unroll
        for (int v = 0; v < 8; ++v) {
            sP[(v + kh * 8) * 40 + lm]      = (_Float16)sc0[v];
            sP[(v + kh * 8) * 40 + 16 + lm] = (_Float16)sc1[v];
        }
        __syncthreads();
        AFrag ap;
        ap.h[0] = *(const h8*)(sP + lm * 40 + kh * 16);
        ap.h[1] = *(const h8*)(sP + lm * 40 + kh * 16 + 8);
        // ---- O += P(16x32) @ V(32x64), V stored transposed [d][s] ----
        {
            const _Float16* vp = vbase + (size_t)(0 * 16 + lm) * SPAD + kt + kh * 16;
            AFrag bv; bv.h[0] = *(const h8*)vp; bv.h[1] = *(const h8*)(vp + 8);
            o0 = wmma_f16(ap.v, bv.v, o0);
        }
        {
            const _Float16* vp = vbase + (size_t)(1 * 16 + lm) * SPAD + kt + kh * 16;
            AFrag bv; bv.h[0] = *(const h8*)vp; bv.h[1] = *(const h8*)(vp + 8);
            o1 = wmma_f16(ap.v, bv.v, o1);
        }
        {
            const _Float16* vp = vbase + (size_t)(2 * 16 + lm) * SPAD + kt + kh * 16;
            AFrag bv; bv.h[0] = *(const h8*)vp; bv.h[1] = *(const h8*)(vp + 8);
            o2 = wmma_f16(ap.v, bv.v, o2);
        }
        {
            const _Float16* vp = vbase + (size_t)(3 * 16 + lm) * SPAD + kt + kh * 16;
            AFrag bv; bv.h[0] = *(const h8*)vp; bv.h[1] = *(const h8*)(vp + 8);
            o3 = wmma_f16(ap.v, bv.v, o3);
        }
        __syncthreads();
    }
    // ---- normalize + store into o[b*S+s][h*64+d] (f16) ----
#pragma unroll
    for (int v = 0; v < 8; ++v) {
        int srow = qrow0 + v + kh * 8;
        if (srow < Ss) {
            float inv = 1.0f / lrow[v];
            _Float16* op = o16 + ((size_t)(b * Ss + srow)) * Ee + hh * HDn + lm;
            op[0]  = (_Float16)(o0[v] * inv);
            op[16] = (_Float16)(o1[v] * inv);
            op[32] = (_Float16)(o2[v] * inv);
            op[48] = (_Float16)(o3[v] * inv);
        }
    }
}

// ---------------- final LN + head1(ReLU) + head2 on CLS rows ----------------
__global__ __launch_bounds__(128) void head_kernel(
    const float* __restrict__ h, const float* __restrict__ nw,
    const float* __restrict__ nb, const float* __restrict__ h1w,
    const float* __restrict__ h1b, const float* __restrict__ h2w,
    const float* __restrict__ h2b, float* __restrict__ out)
{
    __shared__ float xln[Ee];
    __shared__ float y1[Ee];
    __shared__ float red[8];
    int b = blockIdx.x, t = threadIdx.x;
    const float* xr = h + (size_t)b * Ss * Ee;   // row 0 = CLS
    float v0 = xr[t], v1 = xr[t + 128], v2 = xr[t + 256];
    float s  = redsum32(v0 + v1 + v2);
    float s2 = redsum32(v0 * v0 + v1 * v1 + v2 * v2);
    int wave = t >> 5, lane = t & 31;
    if (lane == 0) { red[wave] = s; red[4 + wave] = s2; }
    __syncthreads();
    s  = red[0] + red[1] + red[2] + red[3];
    s2 = red[4] + red[5] + red[6] + red[7];
    float mu  = s * (1.0f / Ee);
    float var = s2 * (1.0f / Ee) - mu * mu;
    float rs  = rsqrtf(var + 1e-5f);
    xln[t]       = (v0 - mu) * rs * nw[t]       + nb[t];
    xln[t + 128] = (v1 - mu) * rs * nw[t + 128] + nb[t + 128];
    xln[t + 256] = (v2 - mu) * rs * nw[t + 256] + nb[t + 256];
    __syncthreads();
    for (int n = t; n < Ee; n += 128) {
        float acc = h1b[n];
        const float* wr = h1w + (size_t)n * Ee;
        for (int e = 0; e < Ee; ++e) acc += xln[e] * wr[e];
        y1[n] = fmaxf(acc, 0.0f);
    }
    __syncthreads();
    if (t < NCn) {
        float acc = h2b[t];
        const float* wr = h2w + (size_t)t * Ee;
        for (int e = 0; e < Ee; ++e) acc += y1[e] * wr[e];
        out[b * NCn + t] = acc;
    }
}

// ---------------- host launch ----------------
extern "C" void kernel_launch(void* const* d_in, const int* in_sizes, int n_in,
                              void* d_out, int out_size, void* d_ws, size_t ws_size,
                              hipStream_t stream) {
    const float* x        = (const float*)d_in[0];
    const float* conv_w   = (const float*)d_in[1];
    const float* conv_b   = (const float*)d_in[2];
    const float* cls_tok  = (const float*)d_in[3];
    const float* ln1_w    = (const float*)d_in[4];
    const float* ln1_b    = (const float*)d_in[5];
    const float* qkv_w    = (const float*)d_in[6];
    const float* proj_w   = (const float*)d_in[7];
    const float* proj_b   = (const float*)d_in[8];
    const float* ln2_w    = (const float*)d_in[9];
    const float* ln2_b    = (const float*)d_in[10];
    const float* fc1_w    = (const float*)d_in[11];
    const float* fc1_b    = (const float*)d_in[12];
    const float* fc2_w    = (const float*)d_in[13];
    const float* fc2_b    = (const float*)d_in[14];
    const float* norm_w   = (const float*)d_in[15];
    const float* norm_b   = (const float*)d_in[16];
    const float* head1_w  = (const float*)d_in[17];
    const float* head1_b  = (const float*)d_in[18];
    const float* head2_w  = (const float*)d_in[19];
    const float* head2_b  = (const float*)d_in[20];

    char* ws = (char*)d_ws;
    // 256-byte aligned workspace layout (total ~33.7 MiB)
    float*    hbuf  = (float*)(ws + 0);                               // 4160*384 f32
    _Float16* ybuf  = (_Float16*)(ws + 6389760);                      // 4160*384 f16
    _Float16* mbuf  = (_Float16*)(ws + 6389760 + 3194880);            // 4160*1536 f16
    _Float16* obuf  = (_Float16*)(ws + 6389760 + 3194880 + 12779520); // 4160*384 f16
    _Float16* qbuf  = (_Float16*)(ws + 25559040);                     // 4*6*1056*64 f16
    _Float16* kbuf  = (_Float16*)(ws + 28803072);
    _Float16* vtbuf = (_Float16*)(ws + 32047104);

    patch_embed<<<6144, 256, 0, stream>>>(x, conv_w, conv_b, hbuf);
    cls_pe<<<3, 128, 0, stream>>>(cls_tok, hbuf);

    const int NQB = (Ss + 15) / 16;   // 65 q-row blocks
    for (int l = 0; l < DEPTHn; ++l) {
        ln_kernel<<<MPAD, 128, 0, stream>>>(hbuf, ln1_w + l * Ee, ln1_b + l * Ee, ybuf);
        gemm_f16w32<EPI_QKV><<<dim3(18, MPAD / 64), 128, 0, stream>>>(
            ybuf, qkv_w + (size_t)l * 3 * Ee * Ee, nullptr, Ee, 3 * Ee,
            nullptr, nullptr, qbuf, kbuf, vtbuf);
        attn_kernel<<<dim3(NQB, Bn * Hh), 32, 0, stream>>>(qbuf, kbuf, vtbuf, obuf);
        gemm_f16w32<EPI_RESID><<<dim3(6, MPAD / 64), 128, 0, stream>>>(
            obuf, proj_w + (size_t)l * Ee * Ee, proj_b + l * Ee, Ee, Ee,
            hbuf, nullptr, nullptr, nullptr, nullptr);
        ln_kernel<<<MPAD, 128, 0, stream>>>(hbuf, ln2_w + l * Ee, ln2_b + l * Ee, ybuf);
        gemm_f16w32<EPI_GELU><<<dim3(24, MPAD / 64), 128, 0, stream>>>(
            ybuf, fc1_w + (size_t)l * MLPD * Ee, fc1_b + l * MLPD, Ee, MLPD,
            nullptr, mbuf, nullptr, nullptr, nullptr);
        gemm_f16w32<EPI_RESID><<<dim3(6, MPAD / 64), 128, 0, stream>>>(
            mbuf, fc2_w + (size_t)l * Ee * MLPD, fc2_b + l * Ee, MLPD, Ee,
            hbuf, nullptr, nullptr, nullptr, nullptr);
    }

    head_kernel<<<Bn, 128, 0, stream>>>(hbuf, norm_w, norm_b,
                                        head1_w, head1_b, head2_w, head2_b,
                                        (float*)d_out);
}
